// EdgeFluxModule_15917148799342
// MI455X (gfx1250) — matrix operand order repaired and checked by reading, not compiled, not run on hardware
//
#include <hip/hip_runtime.h>
#include <hip/hip_bf16.h>

// CDNA5 (gfx1250) wave32 + WMMA implementation of the EdgeFlux GNN.
// Optimization: phi1(hi)+phi1(hj)+phi2(hi)+phi2(hj) == g(hi)+g(hj) with
// g = phi1+phi2 computed per NODE (100k) instead of per EDGE (1.2M).
// All GEMMs via v_wmma_f32_16x16x32_f16; weights pre-swizzled to B-fragment
// layout in LDS; activations staged through LDS for C->A layout transpose.

typedef __attribute__((ext_vector_type(16))) _Float16 v16h;
typedef __attribute__((ext_vector_type(8)))  float    v8f;

union AFrag { _Float16 h[16]; v16h v; uint4 q[2]; };
union H8    { uint4 q; _Float16 h[8]; };

__device__ __forceinline__ float gelu_f(float x) {
    // exact (erf-based) gelu, matching approximate=False
    return 0.5f * x * (1.0f + erff(x * 0.70710678118654752440f));
}

__device__ __forceinline__ v8f wmma_f16(v16h a, v16h b, v8f c) {
    return __builtin_amdgcn_wmma_f32_16x16x32_f16(false, a, false, b, (short)0, c,
                                                  false, false);
}

// B fragment: pre-swizzled in LDS, 32 lanes x 16 halfs contiguous per fragment.
__device__ __forceinline__ v16h lds_B(const _Float16* frags, int idx) {
    int lane = threadIdx.x & 31;
    return *(const v16h*)(frags + idx * 512 + lane * 16);
}

// A fragment from row-major f16 activation tile in LDS (16 rows x stride).
// Lane l: M = l&15 ; halfs 0..7 -> K = kt*32 + kb + j ; halfs 8..15 -> +16.
__device__ __forceinline__ v16h lds_A(const _Float16* act, int stride, int kt) {
    int lane = threadIdx.x & 31;
    int m = lane & 15;
    int kb = kt * 32 + ((lane & 16) ? 8 : 0);
    const _Float16* p = act + m * stride + kb;
    AFrag af;
    af.q[0] = *(const uint4*)(p);
    af.q[1] = *(const uint4*)(p + 16);
    return af.v;
}

// Store a 16x16 f32 C tile (+bias, optional gelu) to row-major f16 LDS tile.
__device__ __forceinline__ void store_C_f16(_Float16* act, int stride, int nt, v8f c,
                                            const float* bias16, bool do_gelu) {
    int lane = threadIdx.x & 31;
    int n = lane & 15;
    int mb = (lane & 16) ? 8 : 0;
    float b = bias16[n];
#pragma unroll
    for (int r = 0; r < 8; ++r) {
        float x = c[r] + b;
        if (do_gelu) x = gelu_f(x);
        act[(mb + r) * stride + nt * 16 + n] = (_Float16)x;
    }
}

// Swizzle f32 weight matrix W[K][N] (row-major) into LDS B fragments (f16).
// Fragment (kt,ntl) -> frag index kt*ntilesTotal + ntOff + ntl.
// Within fragment: lane l half j holds W[kt*32 + (l>>4)*8 + j + (j>=8 ? 8 : 0)][ntl*16 + (l&15)].
__device__ void stage_w(_Float16* fragBase, const float* W, int Kreal, int Kpad,
                        int N, int ntilesTotal, int ntOff) {
    int ktiles = Kpad >> 5, ntiles = N >> 4;
    int pairs = ktiles * ntiles * 32;
    for (int p = threadIdx.x; p < pairs; p += blockDim.x) {
        int l = p & 31, f = p >> 5;
        int kt = f / ntiles, ntl = f - kt * ntiles;
        int krow = kt * 32 + ((l & 16) ? 8 : 0);
        int col = ntl * 16 + (l & 15);
        _Float16* o = fragBase + ((kt * ntilesTotal + ntOff + ntl) * 32 + l) * 16;
#pragma unroll
        for (int j = 0; j < 16; ++j) {
            int k = krow + j + ((j >= 8) ? 8 : 0);
            o[j] = (_Float16)((k < Kreal) ? W[k * N + col] : 0.0f);
        }
    }
}

// A fragment = f16(g[src] + g[dst]) gathered from global (rows of 64 floats).
__device__ __forceinline__ v16h gather_v(const float* g, int s, int d, int kOff) {
    const float4* qs = (const float4*)(g + (long)s * 64 + kOff);
    const float4* qd = (const float4*)(g + (long)d * 64 + kOff);
    float4 s0 = qs[0], s1 = qs[1], s2 = qs[4], s3 = qs[5];
    float4 d0 = qd[0], d1 = qd[1], d2 = qd[4], d3 = qd[5];
    float a0[8] = {s0.x + d0.x, s0.y + d0.y, s0.z + d0.z, s0.w + d0.w,
                   s1.x + d1.x, s1.y + d1.y, s1.z + d1.z, s1.w + d1.w};
    float a1[8] = {s2.x + d2.x, s2.y + d2.y, s2.z + d2.z, s2.w + d2.w,
                   s3.x + d3.x, s3.y + d3.y, s3.z + d3.z, s3.w + d3.w};
    AFrag af;
#pragma unroll
    for (int j = 0; j < 8; ++j) {
        af.h[j] = (_Float16)a0[j];
        af.h[8 + j] = (_Float16)a1[j];
    }
    return af.v;
}

// ------------------------- node kernel: g = phi1(h)+phi2(h) ------------------
#define F_WN1 0    // 4 frags  (5->64, K padded to 32)
#define F_WN2 4    // 8 frags  (64->64)
#define F_WP1A 12  // 4 frags  (64->32)
#define F_WP1B 16  // 4 frags  (32->64)
#define F_WP2A 20  // 4 frags
#define F_WP2B 24  // 4 frags
#define NODE_FRAGS 28
#define NODE_SMEM (NODE_FRAGS * 1024 + 4 * 4096)

__global__ void node_g_kernel(const float* __restrict__ u,
    const float* wN1, const float* bN1, const float* wN2, const float* bN2,
    const float* wP1A, const float* bP1A, const float* wP1B, const float* bP1B,
    const float* wP2A, const float* bP2A, const float* wP2B, const float* bP2B,
    float* __restrict__ gOut, int ntiles) {
    extern __shared__ char smem[];
    _Float16* frags = (_Float16*)smem;
    stage_w(frags + F_WN1 * 512, wN1, 5, 32, 64, 4, 0);
    stage_w(frags + F_WN2 * 512, wN2, 64, 64, 64, 4, 0);
    stage_w(frags + F_WP1A * 512, wP1A, 64, 64, 32, 2, 0);
    stage_w(frags + F_WP1B * 512, wP1B, 32, 32, 64, 4, 0);
    stage_w(frags + F_WP2A * 512, wP2A, 64, 64, 32, 2, 0);
    stage_w(frags + F_WP2B * 512, wP2B, 32, 32, 64, 4, 0);
    __syncthreads();

    int wid = threadIdx.x >> 5, lane = threadIdx.x & 31;
    char* wb = smem + NODE_FRAGS * 1024 + wid * 4096;
    _Float16* bufA = (_Float16*)wb;            // 16x64 staging
    _Float16* bufH = (_Float16*)(wb + 2048);   // 16x64 h
    int wpb = blockDim.x >> 5;
    int gw = blockIdx.x * wpb + wid;
    int nW = gridDim.x * wpb;
    int m = lane & 15, kb = (lane & 16) ? 8 : 0;

    for (int t = gw; t < ntiles; t += nW) {
        long nb = (long)t * 16;
        // phi_node layer1 (K=5 zero-padded to 32), gelu
        AFrag au;
#pragma unroll
        for (int j = 0; j < 8; ++j) {
            int k0 = kb + j;
            au.h[j] = (k0 < 5) ? (_Float16)u[(nb + m) * 5 + k0] : (_Float16)0.0f;
            au.h[8 + j] = (_Float16)0.0f;
        }
#pragma unroll
        for (int nt = 0; nt < 4; ++nt) {
            v8f c = {};
            c = wmma_f16(au.v, lds_B(frags, F_WN1 + nt), c);
            store_C_f16(bufA, 64, nt, c, bN1 + nt * 16, true);
        }
        // phi_node layer2 -> h (no act)
        {
            v16h a0 = lds_A(bufA, 64, 0), a1 = lds_A(bufA, 64, 1);
#pragma unroll
            for (int nt = 0; nt < 4; ++nt) {
                v8f c = {};
                c = wmma_f16(a0, lds_B(frags, F_WN2 + nt), c);
                c = wmma_f16(a1, lds_B(frags, F_WN2 + 4 + nt), c);
                store_C_f16(bufH, 64, nt, c, bN2 + nt * 16, false);
            }
        }
        v8f cg[4] = {};
        {
            v16h h0 = lds_A(bufH, 64, 0), h1 = lds_A(bufH, 64, 1);
            // phi1 layer1, gelu
#pragma unroll
            for (int nt = 0; nt < 2; ++nt) {
                v8f c = {};
                c = wmma_f16(h0, lds_B(frags, F_WP1A + nt), c);
                c = wmma_f16(h1, lds_B(frags, F_WP1A + 2 + nt), c);
                store_C_f16(bufA, 32, nt, c, bP1A + nt * 16, true);
            }
            {   // phi1 layer2 accumulate into g
                v16h a = lds_A(bufA, 32, 0);
#pragma unroll
                for (int nt = 0; nt < 4; ++nt)
                    cg[nt] = wmma_f16(a, lds_B(frags, F_WP1B + nt), cg[nt]);
            }
            // phi2 layer1, gelu
#pragma unroll
            for (int nt = 0; nt < 2; ++nt) {
                v8f c = {};
                c = wmma_f16(h0, lds_B(frags, F_WP2A + nt), c);
                c = wmma_f16(h1, lds_B(frags, F_WP2A + 2 + nt), c);
                store_C_f16(bufA, 32, nt, c, bP2A + nt * 16, true);
            }
            {   // phi2 layer2 accumulate into g
                v16h a = lds_A(bufA, 32, 0);
#pragma unroll
                for (int nt = 0; nt < 4; ++nt)
                    cg[nt] = wmma_f16(a, lds_B(frags, F_WP2B + nt), cg[nt]);
            }
        }
        // store g (f32) with both layer-2 biases
        int n = lane & 15, mb = (lane & 16) ? 8 : 0;
#pragma unroll
        for (int nt = 0; nt < 4; ++nt) {
            float bs = bP1B[nt * 16 + n] + bP2B[nt * 16 + n];
#pragma unroll
            for (int r = 0; r < 8; ++r)
                gOut[(nb + mb + r) * 64 + nt * 16 + n] = cg[nt][r] + bs;
        }
    }
}

// ------------------------------ edge kernel ---------------------------------
#define F_WE2 0    // 2 frags  (32->32, phi_edge layer2)
#define F_WM1 2    // 24 frags (96->128)
#define F_WM2 26   // 16 frags (128->64)
#define F_WH 42    // 24 frags (64->192 = [psi_rho|psi_e|psi_rhou] hidden)
#define EDGE_FRAGS 66
#define EDGE_PW 11776
#define EDGE_SMEM (EDGE_FRAGS * 1024 + 4 * EDGE_PW)

__global__ void edge_flux_kernel(
    const float* __restrict__ g, const long long* __restrict__ eidx,
    const float* __restrict__ attr,
    const float* wE1, const float* bE1, const float* wE2, const float* bE2,
    const float* wM1, const float* bM1, const float* wM2, const float* bM2,
    const float* wR0, const float* bR0, const float* wR1, const float* bR1,
    const float* wE0, const float* bE0, const float* wEf, const float* bEf,
    const float* wU0, const float* bU0, const float* wU1, const float* bU1,
    float* __restrict__ out, long NEl, int ntiles) {
    extern __shared__ char smem[];
    _Float16* frags = (_Float16*)smem;
    stage_w(frags + F_WE2 * 512, wE2, 32, 32, 32, 2, 0);
    stage_w(frags + F_WM1 * 512, wM1, 96, 96, 128, 8, 0);
    stage_w(frags + F_WM2 * 512, wM2, 128, 128, 64, 4, 0);
    stage_w(frags + F_WH * 512, wR0, 64, 64, 64, 12, 0);
    stage_w(frags + F_WH * 512, wE0, 64, 64, 64, 12, 4);
    stage_w(frags + F_WH * 512, wU0, 64, 64, 64, 12, 8);
    __syncthreads();

    int wid = threadIdx.x >> 5, lane = threadIdx.x & 31;
    char* wb = smem + EDGE_FRAGS * 1024 + wid * EDGE_PW;
    _Float16* eps16 = (_Float16*)(wb);           // 16x32
    _Float16* act16 = (_Float16*)(wb + 1024);    // 16x128 (reused as 16x64 m2)
    _Float16* hid16 = (_Float16*)(wb + 5120);    // 16x192
    int* esrc = (int*)(wb + 11264);
    int* edst = (int*)(wb + 11328);
    float* ex = (float*)(wb + 11392);
    float* ey = (float*)(wb + 11456);
    float* er = (float*)(wb + 11520);

    int wpb = blockDim.x >> 5;
    int gw = blockIdx.x * wpb + wid;
    int nW = gridDim.x * wpb;
    int m = lane & 15, kb = (lane & 16) ? 8 : 0;

    for (int t = gw; t < ntiles; t += nW) {
        long eb = (long)t * 16;
        if (lane < 16) {
            long e = eb + lane;
            esrc[lane] = (int)eidx[e];
            edst[lane] = (int)eidx[NEl + e];
            ex[lane] = attr[e * 3 + 0];
            ey[lane] = attr[e * 3 + 1];
            er[lane] = attr[e * 3 + 2];
            long e2 = eb + (long)nW * 16 + lane;   // prefetch next tile
            if (e2 < NEl) {
                __builtin_prefetch(&eidx[e2], 0, 1);
                __builtin_prefetch(&attr[e2 * 3], 0, 1);
            }
        }
        asm volatile("s_wait_dscnt 0x0" ::: "memory");

        // phi_edge: layer1 (1->32 + gelu) built directly in A layout, layer2 WMMA
        float r_m = er[m];
        AFrag aE;
#pragma unroll
        for (int j = 0; j < 8; ++j) {
            int k0 = kb + j, k1 = 16 + kb + j;
            aE.h[j] = (_Float16)gelu_f(r_m * wE1[k0] + bE1[k0]);
            aE.h[8 + j] = (_Float16)gelu_f(r_m * wE1[k1] + bE1[k1]);
        }
#pragma unroll
        for (int nt = 0; nt < 2; ++nt) {
            v8f c = {};
            c = wmma_f16(aE.v, lds_B(frags, F_WE2 + nt), c);
            store_C_f16(eps16, 32, nt, c, bE2 + nt * 16, false);
        }

        // phi_msg layer1: K=96 (v: gathered g[src]+g[dst]; eps from LDS)
        int s = esrc[m], d = edst[m];
        v8f cm[8] = {};
#pragma unroll
        for (int kt = 0; kt < 2; ++kt) {
            v16h a = gather_v(g, s, d, kt * 32 + kb);
#pragma unroll
            for (int nt = 0; nt < 8; ++nt)
                cm[nt] = wmma_f16(a, lds_B(frags, F_WM1 + kt * 8 + nt), cm[nt]);
        }
        {
            v16h a = lds_A(eps16, 32, 0);
#pragma unroll
            for (int nt = 0; nt < 8; ++nt)
                cm[nt] = wmma_f16(a, lds_B(frags, F_WM1 + 16 + nt), cm[nt]);
        }
#pragma unroll
        for (int nt = 0; nt < 8; ++nt)
            store_C_f16(act16, 128, nt, cm[nt], bM1 + nt * 16, true);

        // phi_msg layer2 -> m_ij (no act)
        v8f c2[4] = {};
#pragma unroll
        for (int kt = 0; kt < 4; ++kt) {
            v16h a = lds_A(act16, 128, kt);
#pragma unroll
            for (int nt = 0; nt < 4; ++nt)
                c2[nt] = wmma_f16(a, lds_B(frags, F_WM2 + kt * 4 + nt), c2[nt]);
        }
#pragma unroll
        for (int nt = 0; nt < 4; ++nt)
            store_C_f16(act16, 64, nt, c2[nt], bM2 + nt * 16, false);

        // heads hidden: 64 -> 192 (= rho|e|rhou), gelu
        {
            v16h a0 = lds_A(act16, 64, 0), a1 = lds_A(act16, 64, 1);
#pragma unroll
            for (int nt = 0; nt < 12; ++nt) {
                v8f c = {};
                c = wmma_f16(a0, lds_B(frags, F_WH + nt), c);
                c = wmma_f16(a1, lds_B(frags, F_WH + 12 + nt), c);
                const float* bp = (nt < 4) ? (bR0 + nt * 16)
                                : (nt < 8) ? (bE0 + (nt - 4) * 16)
                                           : (bU0 + (nt - 8) * 16);
                store_C_f16(hid16, 192, nt, c, bp, true);
            }
        }
        asm volatile("s_wait_dscnt 0x0" ::: "memory");

        // final width-{1,1,2} heads + geometry. lanes 0..15: rho/e, 16..31: rhou
        {
            long e = eb + m;
            float dx = ex[m], dy = ey[m], rr = er[m];
            float inv = 1.0f / (rr + 1e-12f);
            float nx = dx * inv, ny = dy * inv;
            if (lane < 16) {
                const _Float16* h0 = hid16 + m * 192;
                float arho = bR1[0], ae = bEf[0];
#pragma unroll
                for (int c8 = 0; c8 < 8; ++c8) {
                    H8 x; x.q = *(const uint4*)(h0 + c8 * 8);
                    H8 y; y.q = *(const uint4*)(h0 + 64 + c8 * 8);
#pragma unroll
                    for (int j = 0; j < 8; ++j) {
                        arho += (float)x.h[j] * wR1[c8 * 8 + j];
                        ae   += (float)y.h[j] * wEf[c8 * 8 + j];
                    }
                }
                out[0 * NEl + 2 * e]     = ae * nx;    // F_e
                out[0 * NEl + 2 * e + 1] = ae * ny;
                out[2 * NEl + 2 * e]     = arho * nx;  // F_rho
                out[2 * NEl + 2 * e + 1] = arho * ny;
                out[6 * NEl + e] = ae;                 // a_e
                out[7 * NEl + e] = arho;               // a_rho
                out[8 * NEl + 2 * e]     = nx;         // n
                out[8 * NEl + 2 * e + 1] = ny;
                out[10 * NEl + e] = rr;                // r
            } else {
                const _Float16* h2 = hid16 + m * 192 + 128;
                float a0 = bU1[0], a1 = bU1[1];
#pragma unroll
                for (int c8 = 0; c8 < 8; ++c8) {
                    H8 x; x.q = *(const uint4*)(h2 + c8 * 8);
#pragma unroll
                    for (int j = 0; j < 8; ++j) {
                        float hv = (float)x.h[j];
                        a0 += hv * wU1[(c8 * 8 + j) * 2 + 0];
                        a1 += hv * wU1[(c8 * 8 + j) * 2 + 1];
                    }
                }
                out[4 * NEl + 2 * e]     = a0 * nx + a1 * (-ny);  // F_rhou (t = [-ny,nx])
                out[4 * NEl + 2 * e + 1] = a0 * ny + a1 * nx;
            }
        }
    }
}

extern "C" void kernel_launch(void* const* d_in, const int* in_sizes, int n_in,
                              void* d_out, int out_size, void* d_ws, size_t ws_size,
                              hipStream_t stream) {
    // d_in: node_u, edge_index(int64), edge_attr, then params in jax pytree
    // (sorted dict key) order: phi1, phi2, phi_edge, phi_msg, phi_node,
    // psi_e, psi_rho, psi_rhou -- each as [w0,b0,w1,b1].
    const float*     node_u = (const float*)d_in[0];
    const long long* eidx   = (const long long*)d_in[1];
    const float*     attr   = (const float*)d_in[2];
    const float* wP1A = (const float*)d_in[3];  const float* bP1A = (const float*)d_in[4];
    const float* wP1B = (const float*)d_in[5];  const float* bP1B = (const float*)d_in[6];
    const float* wP2A = (const float*)d_in[7];  const float* bP2A = (const float*)d_in[8];
    const float* wP2B = (const float*)d_in[9];  const float* bP2B = (const float*)d_in[10];
    const float* wE1  = (const float*)d_in[11]; const float* bE1  = (const float*)d_in[12];
    const float* wE2  = (const float*)d_in[13]; const float* bE2  = (const float*)d_in[14];
    const float* wM1  = (const float*)d_in[15]; const float* bM1  = (const float*)d_in[16];
    const float* wM2  = (const float*)d_in[17]; const float* bM2  = (const float*)d_in[18];
    const float* wN1  = (const float*)d_in[19]; const float* bN1  = (const float*)d_in[20];
    const float* wN2  = (const float*)d_in[21]; const float* bN2  = (const float*)d_in[22];
    const float* wE0  = (const float*)d_in[23]; const float* bE0  = (const float*)d_in[24];
    const float* wEf  = (const float*)d_in[25]; const float* bEf  = (const float*)d_in[26];
    const float* wR0  = (const float*)d_in[27]; const float* bR0  = (const float*)d_in[28];
    const float* wR1  = (const float*)d_in[29]; const float* bR1  = (const float*)d_in[30];
    const float* wU0  = (const float*)d_in[31]; const float* bU0  = (const float*)d_in[32];
    const float* wU1  = (const float*)d_in[33]; const float* bU1  = (const float*)d_in[34];

    long NNl = (long)in_sizes[0] / 5;
    long NEl = (long)in_sizes[2] / 3;
    float* gbuf = (float*)d_ws;               // NNl x 64 f32 (25.6 MB)
    int nodeTiles = (int)(NNl / 16);
    int edgeTiles = (int)(NEl / 16);

    (void)hipFuncSetAttribute((const void*)node_g_kernel,
                              hipFuncAttributeMaxDynamicSharedMemorySize, NODE_SMEM);
    (void)hipFuncSetAttribute((const void*)edge_flux_kernel,
                              hipFuncAttributeMaxDynamicSharedMemorySize, EDGE_SMEM);

    node_g_kernel<<<160, 128, NODE_SMEM, stream>>>(
        node_u, wN1, bN1, wN2, bN2, wP1A, bP1A, wP1B, bP1B, wP2A, bP2A, wP2B, bP2B,
        gbuf, nodeTiles);
    edge_flux_kernel<<<512, 128, EDGE_SMEM, stream>>>(
        gbuf, eidx, attr, wE1, bE1, wE2, bE2, wM1, bM1, wM2, bM2,
        wR0, bR0, wR1, bR1, wE0, bE0, wEf, bEf, wU0, bU0, wU1, bU1,
        (float*)d_out, NEl, edgeTiles);
}
